// QLoRABigNet_42984032698758
// MI455X (gfx1250) — compile-verified, matrix operand order checked by reading, not run on hardware
//
#include <hip/hip_runtime.h>

typedef __bf16 bf16_t;
typedef __bf16 v16bf __attribute__((ext_vector_type(16)));
typedef float  v8f   __attribute__((ext_vector_type(8)));
typedef unsigned int v4u __attribute__((ext_vector_type(4)));
typedef int    v4i   __attribute__((ext_vector_type(4)));

union FragBF {
    v16bf bf;
    v4u   u[2];
};

#define DMODEL 1024
#define M_TOTAL 32768
#define LDA 40   // padded LDS row stride (elements); 80B keeps 16B alignment

// ---------------------------------------------------------------------------
// gfx1250 async global->LDS path (ASYNCcnt) with guarded fallback.
// Builtin prototype (from compiler diagnostic): param0 = v4i addrspace(1)*,
// param1 = v4i addrspace(3)*, then imm offset, imm cpol.
// ---------------------------------------------------------------------------
#if defined(__HIP_DEVICE_COMPILE__) && __has_builtin(__builtin_amdgcn_global_load_async_to_lds_b128)
#define HAVE_ASYNC_LDS 1
#else
#define HAVE_ASYNC_LDS 0
#endif

typedef __attribute__((address_space(1))) v4i GAS_v4i;
typedef __attribute__((address_space(3))) v4i LAS_v4i;

__device__ __forceinline__ void ld_async_b128(const bf16_t* g, bf16_t* l)
{
#if HAVE_ASYNC_LDS
    __builtin_amdgcn_global_load_async_to_lds_b128((GAS_v4i*)g, (LAS_v4i*)l, 0, 0);
#else
    *(v4u*)l = *(const v4u*)g;
#endif
}

__device__ __forceinline__ void wait_async_all()
{
#if HAVE_ASYNC_LDS
#if __has_builtin(__builtin_amdgcn_s_wait_asynccnt)
    __builtin_amdgcn_s_wait_asynccnt(0);
#else
    asm volatile("s_wait_asynccnt 0x0" ::: "memory");
#endif
#endif
}

// ---------------------------------------------------------------------------
// GEMM: Y[m][n] = sum_k A_bf16[m][k] * Weff_bf16[n][k] + bias[n]
// WG = 256 threads = 8 waves (4M x 2N), WG tile 128(M) x 64(N), K step 32.
// LDS double-buffered; next K-slab loaded async while WMMAs run on current.
// ---------------------------------------------------------------------------
__global__ __launch_bounds__(256)
void gemm_bf16_wmma(float* __restrict__ Y,
                    const bf16_t* __restrict__ A,
                    const bf16_t* __restrict__ W,
                    const float* __restrict__ bias)
{
    __shared__ bf16_t lA[2][128 * LDA];
    __shared__ bf16_t lB[2][64 * LDA];

    const int tid   = threadIdx.x;
    const int lane  = tid & 31;
    const int w     = tid >> 5;
    const int waveM = w >> 1;     // 0..3
    const int waveN = w & 1;      // 0..1
    const int l16   = lane & 15;
    const int hi    = lane >> 4;  // 0 or 1

    const int mBase = blockIdx.y * 128;
    const int nBase = blockIdx.x * 64;

    v8f acc[2][2] = {};

    // staging: one b128 (8 bf16) per lane per chunk
    const int arow0 = tid >> 2;         // A chunks 0..255   -> rows 0..63
    const int arow1 = 64 + (tid >> 2);  // A chunks 256..511 -> rows 64..127
    const int asub  = tid & 3;          // 8-elem group within 32-col slab
    const int brow  = tid >> 2;         // B: 64 rows x 4 chunks
    const int bsub  = tid & 3;

    auto stage = [&](int buf, int kk) {
        ld_async_b128(A + (size_t)(mBase + arow0) * DMODEL + kk + asub * 8,
                      &lA[buf][arow0 * LDA + asub * 8]);
        ld_async_b128(A + (size_t)(mBase + arow1) * DMODEL + kk + asub * 8,
                      &lA[buf][arow1 * LDA + asub * 8]);
        ld_async_b128(W + (size_t)(nBase + brow) * DMODEL + kk + bsub * 8,
                      &lB[buf][brow * LDA + bsub * 8]);
    };

    stage(0, 0);
    wait_async_all();
    __syncthreads();

    int cur = 0;
    for (int kk = 0; kk < DMODEL; kk += 32) {
        const int  nxt  = cur ^ 1;
        const bool more = (kk + 32 < DMODEL);
        if (more) stage(nxt, kk + 32);                 // async fill of next buffer
        if (kk + 64 < DMODEL)                          // L2 warm-up two slabs ahead
            __builtin_prefetch(A + (size_t)(mBase + arow0) * DMODEL + kk + 64, 0, 1);

        // ---- fragments per ISA 16-bit A/B layouts ----
        FragBF a[2], b[2];
#pragma unroll
        for (int sm = 0; sm < 2; ++sm) {
            int r = waveM * 32 + sm * 16 + l16;
            a[sm].u[0] = *(const v4u*)&lA[cur][r * LDA + hi * 8];       // K 0..7 / 8..15
            a[sm].u[1] = *(const v4u*)&lA[cur][r * LDA + 16 + hi * 8];  // K 16..23 / 24..31
        }
#pragma unroll
        for (int sn = 0; sn < 2; ++sn) {
            int c = waveN * 32 + sn * 16 + l16;
            b[sn].u[0] = *(const v4u*)&lB[cur][c * LDA + hi * 8];
            b[sn].u[1] = *(const v4u*)&lB[cur][c * LDA + 16 + hi * 8];
        }

#pragma unroll
        for (int sm = 0; sm < 2; ++sm)
#pragma unroll
            for (int sn = 0; sn < 2; ++sn)
                acc[sm][sn] = __builtin_amdgcn_wmma_f32_16x16x32_bf16(
                    false, a[sm].bf, false, b[sn].bf, (short)0, acc[sm][sn], false, false);

        if (more) wait_async_all();                    // next buffer landed
        __syncthreads();                               // all waves done reading cur
        cur = nxt;
    }

    // ---- epilogue: bias + store (C/D layout: lane<16 -> M=i, lane>=16 -> M=8+i) ----
#pragma unroll
    for (int sm = 0; sm < 2; ++sm) {
#pragma unroll
        for (int sn = 0; sn < 2; ++sn) {
            int n = nBase + waveN * 32 + sn * 16 + l16;
            float bn = bias[n];
#pragma unroll
            for (int i = 0; i < 8; ++i) {
                int m = mBase + waveM * 32 + sm * 16 + hi * 8 + i;
                Y[(size_t)m * DMODEL + n] = acc[sm][sn][i] + bn;
            }
        }
    }
}

// ---------------------------------------------------------------------------
// Per-layer weight prep: Weff[o][k] = (q/7.5 - 1) * scale[o][k/16] + (lb @ la)[o][k]
// ---------------------------------------------------------------------------
__global__ __launch_bounds__(256)
void prep_weight(const int* __restrict__ q, const float* __restrict__ sc,
                 const float* __restrict__ la, const float* __restrict__ lb,
                 bf16_t* __restrict__ W)
{
    int idx = blockIdx.x * 256 + threadIdx.x;   // 0..1048575
    int o = idx >> 10;
    int k = idx & 1023;
    float wv = (float)q[idx] * (1.0f / 7.5f) - 1.0f;
    wv *= sc[(o << 6) + (k >> 4)];
    float accv = 0.0f;
#pragma unroll
    for (int r = 0; r < 32; ++r)
        accv = fmaf(lb[(o << 5) + r], la[(r << 10) + k], accv);
    W[idx] = (bf16_t)(wv + accv);
}

// ---------------------------------------------------------------------------
// Fused LayerNorm + exact GELU, output as bf16 (input to next GEMM).
// One WG (256 threads) per row of 1024 floats; wave32 shuffles + LDS reduce.
// ---------------------------------------------------------------------------
__global__ __launch_bounds__(256)
void ln_gelu(bf16_t* __restrict__ out, const float* __restrict__ Y,
             const float* __restrict__ gamma, const float* __restrict__ beta)
{
    __shared__ float red1[8], red2[8];
    int row = blockIdx.x, tid = threadIdx.x;
    const float4* yr = (const float4*)(Y + (size_t)row * DMODEL);
    float4 v = yr[tid];
    float s = v.x + v.y + v.z + v.w;
    float q = v.x * v.x + v.y * v.y + v.z * v.z + v.w * v.w;
#pragma unroll
    for (int off = 16; off > 0; off >>= 1) {
        s += __shfl_xor(s, off, 32);
        q += __shfl_xor(q, off, 32);
    }
    int wv = tid >> 5, lane = tid & 31;
    if (lane == 0) { red1[wv] = s; red2[wv] = q; }
    __syncthreads();
    if (tid == 0) {
        float S = 0.f, Q = 0.f;
#pragma unroll
        for (int i = 0; i < 8; ++i) { S += red1[i]; Q += red2[i]; }
        red1[0] = S; red2[0] = Q;
    }
    __syncthreads();
    float mean = red1[0] * (1.0f / 1024.0f);
    float var  = red2[0] * (1.0f / 1024.0f) - mean * mean;
    float rstd = rsqrtf(var + 1e-5f);
    const float4 g4 = ((const float4*)gamma)[tid];
    const float4 b4 = ((const float4*)beta)[tid];
    float tv[4] = {v.x, v.y, v.z, v.w};
    float gg[4] = {g4.x, g4.y, g4.z, g4.w};
    float bb[4] = {b4.x, b4.y, b4.z, b4.w};
    bf16_t* op = out + (size_t)row * DMODEL + tid * 4;
#pragma unroll
    for (int j = 0; j < 4; ++j) {
        float t  = (tv[j] - mean) * rstd * gg[j] + bb[j];
        float gl = 0.5f * t * (1.0f + erff(t * 0.70710678118654752f));
        op[j] = (bf16_t)gl;
    }
}

// ---------------------------------------------------------------------------
// Elementwise helpers
// ---------------------------------------------------------------------------
__global__ __launch_bounds__(256)
void copy_f32(float* __restrict__ o, const float* __restrict__ i)
{
    int idx = blockIdx.x * 256 + threadIdx.x;
    ((float4*)o)[idx] = ((const float4*)i)[idx];
}

__global__ __launch_bounds__(256)
void cvt_f32_bf16(bf16_t* __restrict__ out, const float* __restrict__ in)
{
    int idx = blockIdx.x * 256 + threadIdx.x;
    const float4 v = ((const float4*)in)[idx];
    bf16_t* op = out + (size_t)idx * 4;
    op[0] = (bf16_t)v.x; op[1] = (bf16_t)v.y;
    op[2] = (bf16_t)v.z; op[3] = (bf16_t)v.w;
}

__global__ __launch_bounds__(256)
void add_f32(float* __restrict__ o, const float* __restrict__ y)
{
    int idx = blockIdx.x * 256 + threadIdx.x;
    float4 a = ((float4*)o)[idx];
    const float4 b = ((const float4*)y)[idx];
    a.x += b.x; a.y += b.y; a.z += b.z; a.w += b.w;
    ((float4*)o)[idx] = a;
}

// ---------------------------------------------------------------------------
// Orchestration: h lives in d_out (f32 residual). Per block:
//   prep(i0) -> cvt(h->Abf) -> GEMM -> LN+GELU(->Abf) -> prep(i1) -> GEMM -> h += Y
// ---------------------------------------------------------------------------
extern "C" void kernel_launch(void* const* d_in, const int* in_sizes, int n_in,
                              void* d_out, int out_size, void* d_ws, size_t ws_size,
                              hipStream_t stream)
{
    const float* x      = (const float*)d_in[0];
    const int*   qw     = (const int*)d_in[1];
    const float* scales = (const float*)d_in[2];
    const float* biases = (const float*)d_in[3];
    const float* la     = (const float*)d_in[4];
    const float* lb     = (const float*)d_in[5];
    const float* lng    = (const float*)d_in[6];
    const float* lnb    = (const float*)d_in[7];
    float* h = (float*)d_out;

    char* ws = (char*)d_ws;
    bf16_t* Abf  = (bf16_t*)(ws);                              // 64 MB
    float*  Ybuf = (float*)(ws + (size_t)64  * 1024 * 1024);   // 128 MB
    bf16_t* Weff = (bf16_t*)(ws + (size_t)192 * 1024 * 1024);  // 2 MB

    const size_t NE = (size_t)M_TOTAL * DMODEL;      // 33554432
    const int vecBlocks = (int)(NE / 4 / 256);       // 32768

    copy_f32<<<vecBlocks, 256, 0, stream>>>(h, x);

    dim3 ggrid(DMODEL / 64, M_TOTAL / 128, 1);       // (16, 256)
    for (int blk = 0; blk < 3; ++blk) {
        int i0 = 2 * blk, i1 = 2 * blk + 1;
        prep_weight<<<4096, 256, 0, stream>>>(qw + (size_t)i0 * DMODEL * DMODEL,
                                              scales + (size_t)i0 * DMODEL * 64,
                                              la + (size_t)i0 * 32 * DMODEL,
                                              lb + (size_t)i0 * DMODEL * 32, Weff);
        cvt_f32_bf16<<<vecBlocks, 256, 0, stream>>>(Abf, h);
        gemm_bf16_wmma<<<ggrid, 256, 0, stream>>>(Ybuf, Abf, Weff,
                                                  biases + (size_t)i0 * DMODEL);
        ln_gelu<<<M_TOTAL, 256, 0, stream>>>(Abf, Ybuf,
                                             lng + (size_t)blk * DMODEL,
                                             lnb + (size_t)blk * DMODEL);
        prep_weight<<<4096, 256, 0, stream>>>(qw + (size_t)i1 * DMODEL * DMODEL,
                                              scales + (size_t)i1 * DMODEL * 64,
                                              la + (size_t)i1 * 32 * DMODEL,
                                              lb + (size_t)i1 * DMODEL * 32, Weff);
        gemm_bf16_wmma<<<ggrid, 256, 0, stream>>>(Ybuf, Abf, Weff,
                                                  biases + (size_t)i1 * DMODEL);
        add_f32<<<vecBlocks, 256, 0, stream>>>(h, Ybuf);
    }
    (void)in_sizes; (void)n_in; (void)out_size; (void)ws_size;
}
